// HybridReasoningLayer_5712306504126
// MI455X (gfx1250) — matrix operand order repaired and checked by reading, not compiled, not run on hardware
//
#include <hip/hip_runtime.h>
#include <cstddef>
#include <cstdint>

typedef __bf16 bf16;
typedef __attribute__((ext_vector_type(16))) __bf16 v16bf;
typedef __attribute__((ext_vector_type(8)))  __bf16 v8bf;
typedef __attribute__((ext_vector_type(8)))  float  v8f;

#define D_ 256
#define H_ 4
#define HD_ 64
#define B_ 16
#define T_ 512
#define NEGV (-1e9f)
#define SCALE_ 0.125f /* 1/sqrt(64) */

__device__ __forceinline__ float gelu_f(float x) {
  return 0.5f * x * (1.0f + erff(x * 0.70710678118654752f));
}

__device__ __forceinline__ v8f vzero8() {
  v8f z = {0.f, 0.f, 0.f, 0.f, 0.f, 0.f, 0.f, 0.f};
  return z;
}

struct bfrag32 { uint4 lo, hi; };

// 16 contiguous bf16 (16B-aligned) -> B fragment (K = khalf*16 + i)
__device__ __forceinline__ v16bf frag16_contig(const bf16* p) {
  bfrag32 u;
  u.lo = *(const uint4*)(p);
  u.hi = *(const uint4*)(p + 8);
  return __builtin_bit_cast(v16bf, u);
}

// A fragment from a bf16 row: two contiguous 8-elem runs
// (K = kk + khalf*16 + i)
__device__ __forceinline__ v16bf frag_a_bf16(const bf16* row, int khalf, int kk) {
  bfrag32 u;
  u.lo = *(const uint4*)(row + kk + khalf * 16);
  u.hi = *(const uint4*)(row + kk + khalf * 16 + 8);
  return __builtin_bit_cast(v16bf, u);
}

// A fragment from an f32 row (vector loads + pack-convert)
__device__ __forceinline__ v16bf frag_a_f32(const float* row, int khalf, int kk) {
  const float4 a0 = *(const float4*)(row + kk + khalf * 16);
  const float4 a1 = *(const float4*)(row + kk + khalf * 16 + 4);
  const float4 b0 = *(const float4*)(row + kk + khalf * 16 + 8);
  const float4 b1 = *(const float4*)(row + kk + khalf * 16 + 12);
  v16bf r;
  r[0] = (bf16)a0.x;  r[1] = (bf16)a0.y;  r[2] = (bf16)a0.z;  r[3] = (bf16)a0.w;
  r[4] = (bf16)a1.x;  r[5] = (bf16)a1.y;  r[6] = (bf16)a1.z;  r[7] = (bf16)a1.w;
  r[8] = (bf16)b0.x;  r[9] = (bf16)b0.y;  r[10] = (bf16)b0.z; r[11] = (bf16)b0.w;
  r[12] = (bf16)b1.x; r[13] = (bf16)b1.y; r[14] = (bf16)b1.z; r[15] = (bf16)b1.w;
  return r;
}

// ---------------------------------------------------------------------------
// Generic tiled GEMM: epi(A[M,K] @ W[K,N] + bias), bf16 WMMA, f32 acc.
// block = 256 (8 waves), tile = 128(M) x 64(N), K-step 32.
// B tile staged TRANSPOSED in LDS so fragments are 128-bit DS loads.
// OMODE: 0 = f32 C (+opt residual), 1 = bf16 C, 2 = bf16 transposed [b][h][d][t]
// ---------------------------------------------------------------------------
template <int ACT, bool RES, int OMODE>
__global__ __launch_bounds__(256) void gemm_kernel(
    const float* __restrict__ A, const float* __restrict__ W,
    const float* __restrict__ bias, const float* __restrict__ resid,
    float* __restrict__ C, bf16* __restrict__ Cb, bf16* __restrict__ Ct,
    int M, int N, int K) {
  __shared__ __align__(16) bf16 As[128][40];
  __shared__ __align__(16) bf16 Bs[64][32];  // [n-in-tile][k] transposed
  const int tid = threadIdx.x;
  const int m0 = blockIdx.y * 128;
  const int n0 = blockIdx.x * 64;
  const int wave = tid >> 5;
  const int lane = tid & 31;
  const int col16 = lane & 15;
  const int khalf = lane >> 4;

  v8f acc[4];
#pragma unroll
  for (int j = 0; j < 4; ++j) acc[j] = vzero8();

  for (int kk = 0; kk < K; kk += 32) {
    __syncthreads();
#pragma unroll
    for (int e = 0; e < 16; ++e) {  // A 128x32, coalesced reads
      int idx = e * 256 + tid;
      int r = idx >> 5, c = idx & 31;
      float vA = (m0 + r < M) ? A[(size_t)(m0 + r) * K + kk + c] : 0.0f;
      As[r][c] = (bf16)vA;
    }
#pragma unroll
    for (int e = 0; e < 8; ++e) {  // B 32x64, store transposed
      int idx = e * 256 + tid;
      int r = idx >> 6, c = idx & 63;
      Bs[c][r] = (bf16)W[(size_t)(kk + r) * N + n0 + c];
    }
    __syncthreads();

    v16bf a = frag_a_bf16(&As[wave * 16 + col16][0], khalf, 0);
#pragma unroll
    for (int nt = 0; nt < 4; ++nt) {
      v16bf b = frag16_contig(&Bs[nt * 16 + col16][khalf * 16]);
      acc[nt] = __builtin_amdgcn_wmma_f32_16x16x32_bf16(
          false, a, false, b, (short)0, acc[nt], false, false);
    }
  }

#pragma unroll
  for (int nt = 0; nt < 4; ++nt) {
#pragma unroll
    for (int i = 0; i < 8; ++i) {
      int r = m0 + wave * 16 + (khalf ? i + 8 : i);
      int c = n0 + nt * 16 + col16;
      if (r < M) {
        float v = acc[nt][i] + bias[c];
        if (ACT == 1) v = gelu_f(v);
        if (OMODE == 0) {
          if (RES) v += resid[(size_t)r * N + c];
          C[(size_t)r * N + c] = v;
        } else if (OMODE == 1) {
          Cb[(size_t)r * N + c] = (bf16)v;
        } else {
          int bb = r >> 9, tl = r & (T_ - 1);
          int hh = c >> 6, dd = c & 63;
          Ct[(((size_t)bb * H_ + hh) * HD_ + dd) * T_ + tl] = (bf16)v;
        }
      }
    }
  }
}

// ---------------------------------------------------------------------------
// Causal flash attention: one wave per (q-tile of 16, head, batch).
// q,k are bf16 [B*T, D] head-major; vt is bf16 V^T [b][h][d][t]; o is f32.
// ---------------------------------------------------------------------------
__global__ __launch_bounds__(32) void attn_kernel(
    const bf16* __restrict__ q, const bf16* __restrict__ k,
    const bf16* __restrict__ vt, float* __restrict__ o) {
  __shared__ __align__(16) bf16 Ps[16][40];
  const int lane = threadIdx.x;
  const int col16 = lane & 15;
  const int khalf = lane >> 4;
  const int qt = blockIdx.x;
  const int h = blockIdx.y;
  const int b = blockIdx.z;
  const bf16* qrow = q + ((size_t)(b * T_) + qt * 16 + col16) * D_ + h * HD_;
  const bf16* kp = k + (size_t)(b * T_) * D_ + h * HD_;
  const bf16* vtp = vt + ((size_t)b * H_ + h) * HD_ * T_;
  float* op = o + ((size_t)(b * T_) + qt * 16) * D_ + h * HD_;

  const v16bf aQ0 = frag_a_bf16(qrow, khalf, 0);
  const v16bf aQ1 = frag_a_bf16(qrow, khalf, 32);

  v8f oacc[4];
#pragma unroll
  for (int nt = 0; nt < 4; ++nt) oacc[nt] = vzero8();
  float mrow[8], lrow[8];
#pragma unroll
  for (int i = 0; i < 8; ++i) { mrow[i] = -3.0e38f; lrow[i] = 0.f; }

  for (int jt = 0; jt <= qt; jt += 2) {
    const bool have2 = (jt + 1) <= qt;
    v8f s0 = vzero8(), s1 = vzero8();
#pragma unroll
    for (int c = 0; c < 2; ++c) {
      const v16bf aC = (c == 0) ? aQ0 : aQ1;
      v16bf bK = frag16_contig(kp + (size_t)(jt * 16 + col16) * D_ + c * 32 + khalf * 16);
      s0 = __builtin_amdgcn_wmma_f32_16x16x32_bf16(false, aC, false, bK,
                                                   (short)0, s0, false, false);
      if (have2) {
        v16bf bK2 = frag16_contig(kp + (size_t)((jt + 1) * 16 + col16) * D_ + c * 32 + khalf * 16);
        s1 = __builtin_amdgcn_wmma_f32_16x16x32_bf16(false, aC, false, bK2,
                                                     (short)0, s1, false, false);
      }
    }
    float scf[8];
#pragma unroll
    for (int i = 0; i < 8; ++i) {
      int rq = qt * 16 + (khalf ? i + 8 : i);
      int c0 = jt * 16 + col16;
      int c1 = c0 + 16;
      float v0 = (c0 <= rq) ? s0[i] * SCALE_ : NEGV;
      float v1 = (have2 && (c1 <= rq)) ? s1[i] * SCALE_ : NEGV;
      float tm = fmaxf(v0, v1);
      tm = fmaxf(tm, __shfl_xor(tm, 8, 16));
      tm = fmaxf(tm, __shfl_xor(tm, 4, 16));
      tm = fmaxf(tm, __shfl_xor(tm, 2, 16));
      tm = fmaxf(tm, __shfl_xor(tm, 1, 16));
      float mn = fmaxf(mrow[i], tm);
      float sc = expf(mrow[i] - mn);
      float p0 = expf(v0 - mn);
      float p1 = expf(v1 - mn);
      float ts = p0 + p1;
      ts += __shfl_xor(ts, 8, 16);
      ts += __shfl_xor(ts, 4, 16);
      ts += __shfl_xor(ts, 2, 16);
      ts += __shfl_xor(ts, 1, 16);
      lrow[i] = lrow[i] * sc + ts;
      mrow[i] = mn;
      scf[i] = sc;
      int row = khalf ? i + 8 : i;
      Ps[row][col16] = (bf16)p0;
      Ps[row][16 + col16] = (bf16)p1;
    }
#pragma unroll
    for (int nt = 0; nt < 4; ++nt)
#pragma unroll
      for (int i = 0; i < 8; ++i) oacc[nt][i] *= scf[i];
    __syncthreads();
    v16bf aP = frag_a_bf16(&Ps[col16][0], khalf, 0);
#pragma unroll
    for (int nt = 0; nt < 4; ++nt) {
      v16bf bV = frag16_contig(vtp + (size_t)(nt * 16 + col16) * T_ + jt * 16 + khalf * 16);
      oacc[nt] = __builtin_amdgcn_wmma_f32_16x16x32_bf16(
          false, aP, false, bV, (short)0, oacc[nt], false, false);
    }
    __syncthreads();
  }

#pragma unroll
  for (int nt = 0; nt < 4; ++nt)
#pragma unroll
    for (int i = 0; i < 8; ++i) {
      int row = khalf ? i + 8 : i;
      op[(size_t)row * D_ + nt * 16 + col16] = oacc[nt][i] / lrow[i];
    }
}

// ---------------------------------------------------------------------------
__global__ __launch_bounds__(256) void ln_kernel(const float* __restrict__ in,
                                                 const float* __restrict__ g,
                                                 const float* __restrict__ bb,
                                                 float* __restrict__ out) {
  __shared__ float red[256];
  __shared__ float sMean, sRstd;
  const int row = blockIdx.x;
  const int j = threadIdx.x;
  float vx = in[(size_t)row * 256 + j];
  red[j] = vx;
  __syncthreads();
  for (int s = 128; s > 0; s >>= 1) {
    if (j < s) red[j] += red[j + s];
    __syncthreads();
  }
  if (j == 0) sMean = red[0] * (1.0f / 256.0f);
  __syncthreads();
  float d = vx - sMean;
  red[j] = d * d;
  __syncthreads();
  for (int s = 128; s > 0; s >>= 1) {
    if (j < s) red[j] += red[j + s];
    __syncthreads();
  }
  if (j == 0) sRstd = rsqrtf(red[0] * (1.0f / 256.0f) + 1e-5f);
  __syncthreads();
  out[(size_t)row * 256 + j] = d * sRstd * g[j] + bb[j];
}

__global__ __launch_bounds__(256) void add_guid_kernel(
    const float* __restrict__ x1, const float* __restrict__ guid,
    float* __restrict__ out) {
  size_t idx = (size_t)blockIdx.x * 256 + threadIdx.x;
  int row = (int)(idx >> 8);
  int c = (int)(idx & 255);
  int b = row >> 9;  // row / T_
  out[idx] = x1[idx] + 0.3f * guid[b * 256 + c];
}

// f32 [K,N] -> bf16 transposed [N,K] (one-time; keeps scan B-frags contiguous)
__global__ __launch_bounds__(256) void cvt_t_kernel(const float* __restrict__ s,
                                                    bf16* __restrict__ d,
                                                    int K, int N) {
  int i = blockIdx.x * 256 + threadIdx.x;
  if (i < K * N) {
    int r = i / N, c = i - r * N;
    d[(size_t)c * K + r] = (bf16)s[i];
  }
}

// ---------------------------------------------------------------------------
// Wave-cooperative M=16 GEMM for the scan. Wt is bf16 TRANSPOSED [N][K].
// ---------------------------------------------------------------------------
template <int NT, typename AF, typename EPI>
__device__ __forceinline__ void gemm16(AF afrag, const bf16* __restrict__ Wt,
                                       const float* __restrict__ bias, int N,
                                       int K, EPI epi) {
  const int tid = threadIdx.x;
  const int wave = tid >> 5;
  const int lane = tid & 31;
  const int col16 = lane & 15;
  const int khalf = lane >> 4;
  v8f acc[NT];
#pragma unroll
  for (int j = 0; j < NT; ++j) acc[j] = vzero8();
  for (int kk = 0; kk < K; kk += 32) {
    v16bf a = afrag(col16, khalf, kk);
#pragma unroll
    for (int j = 0; j < NT; ++j) {
      const int nt = wave + j * 8;
      v16bf b = frag16_contig(Wt + (size_t)(nt * 16 + col16) * K + kk + khalf * 16);
      acc[j] = __builtin_amdgcn_wmma_f32_16x16x32_bf16(
          false, a, false, b, (short)0, acc[j], false, false);
    }
  }
#pragma unroll
  for (int j = 0; j < NT; ++j) {
    const int nt = wave + j * 8;
#pragma unroll
    for (int i = 0; i < 8; ++i) {
      const int r = khalf ? i + 8 : i;
      const int c = nt * 16 + col16;
      epi(r, c, acc[j][i] + bias[c]);
    }
  }
}

// ---------------------------------------------------------------------------
// Persistent single-workgroup sequential scan (512 steps, B=16 rows batched
// into the WMMA M dimension). LDS: h f32 16K | comb/delta 16K | {qt,ctxv}/g1 16K
// ---------------------------------------------------------------------------
__global__ __launch_bounds__(256) void scan_kernel(
    const float* __restrict__ xo, const float* __restrict__ h0,
    const bf16* __restrict__ wq, const bf16* __restrict__ wk,
    const bf16* __restrict__ wv, const bf16* __restrict__ wo,
    const bf16* __restrict__ wps, const bf16* __restrict__ wtr1,
    const bf16* __restrict__ wtr2, const float* __restrict__ bq,
    const float* __restrict__ bk, const float* __restrict__ bv,
    const float* __restrict__ bo, const float* __restrict__ bps,
    const float* __restrict__ btr1, const float* __restrict__ btr2,
    const float* __restrict__ ln3g, const float* __restrict__ ln3b,
    const float* __restrict__ ln4g, const float* __restrict__ ln4b,
    float* __restrict__ kbuf, float* __restrict__ vbuf,
    float* __restrict__ hout) {
  __shared__ __align__(16) unsigned char smem[16 * 256 * 4 + 16 * 512 * 2 + 16 * 512 * 2];
  float* hS = (float*)smem;                                    // [16][256] f32
  bf16* combS = (bf16*)(smem + 16 * 256 * 4);                  // [16][512] bf16
  unsigned char* region = smem + 16 * 256 * 4 + 16 * 512 * 2;  // 16KB reused
  bf16* qtS = (bf16*)region;                                   // [16][256]
  bf16* ctxvS = (bf16*)(region + 16 * 256 * 2);                // [16][256]
  bf16* g1S = (bf16*)region;                                   // [16][512]
  float* deltaS = (float*)combS;                               // [16][256] alias

  const int tid = threadIdx.x;
  for (int i = tid; i < 16 * 256; i += 256) hS[i] = h0[i];
  __syncthreads();

  for (int t = 0; t < T_; ++t) {
    auto af_xo = [&](int r, int khalf, int kk) -> v16bf {
      return frag_a_f32(xo + ((size_t)r * T_ + t) * D_, khalf, kk);
    };
    // qt = xt @ ta_wq + bq (only needed for t>0; ctx is zeroed at t==0)
    if (t > 0) {
      gemm16<2>(af_xo, wq, bq, 256, 256,
                [&](int r, int c, float val) { qtS[r * 256 + c] = (bf16)val; });
    }
    __syncthreads();

    // trajectory attention over kbuf/vbuf[0..t): 4 lanes per (b, head)
    if (t > 0) {
      const int pairId = tid >> 2;
      const int sub = tid & 3;
      const int bb2 = pairId >> 2;
      const int hh = pairId & 3;
      float qv[64];
      {
        const uint4* q4 = (const uint4*)(qtS + bb2 * 256 + hh * 64);
#pragma unroll
        for (int jq = 0; jq < 8; ++jq) {
          v8bf q8 = __builtin_bit_cast(v8bf, q4[jq]);
#pragma unroll
          for (int e = 0; e < 8; ++e) qv[jq * 8 + e] = (float)q8[e];
        }
      }
      float ml = -3.0e38f, ll = 0.f;
      float ctx[64];
#pragma unroll
      for (int dd = 0; dd < 64; ++dd) ctx[dd] = 0.f;
      for (int tp = sub; tp < t; tp += 4) {
        const float4* kr4 = (const float4*)(kbuf + ((size_t)tp * 16 + bb2) * 256 + hh * 64);
        const float4* vr4 = (const float4*)(vbuf + ((size_t)tp * 16 + bb2) * 256 + hh * 64);
        float s = 0.f;
#pragma unroll
        for (int j4 = 0; j4 < 16; ++j4) {
          float4 kv = kr4[j4];
          s += qv[4 * j4 + 0] * kv.x + qv[4 * j4 + 1] * kv.y +
               qv[4 * j4 + 2] * kv.z + qv[4 * j4 + 3] * kv.w;
        }
        s *= SCALE_;
        float mn = fmaxf(ml, s);
        float cs = expf(ml - mn);
        float p = expf(s - mn);
        ll = ll * cs + p;
#pragma unroll
        for (int j4 = 0; j4 < 16; ++j4) {
          float4 vv = vr4[j4];
          ctx[4 * j4 + 0] = ctx[4 * j4 + 0] * cs + p * vv.x;
          ctx[4 * j4 + 1] = ctx[4 * j4 + 1] * cs + p * vv.y;
          ctx[4 * j4 + 2] = ctx[4 * j4 + 2] * cs + p * vv.z;
          ctx[4 * j4 + 3] = ctx[4 * j4 + 3] * cs + p * vv.w;
        }
        ml = mn;
      }
      float mg = ml;
      mg = fmaxf(mg, __shfl_xor(mg, 1, 4));
      mg = fmaxf(mg, __shfl_xor(mg, 2, 4));
      float cs2 = expf(ml - mg);
      float lg = ll * cs2;
      lg += __shfl_xor(lg, 1, 4);
      lg += __shfl_xor(lg, 2, 4);
      float inv = 1.0f / lg;
#pragma unroll
      for (int dd = 0; dd < 64; ++dd) {
        float cv = ctx[dd] * cs2;
        cv += __shfl_xor(cv, 1, 4);
        cv += __shfl_xor(cv, 2, 4);
        ctx[dd] = cv * inv;
      }
#pragma unroll
      for (int e = 0; e < 16; ++e) {
        int dd = sub * 16 + e;
        ctxvS[bb2 * 256 + hh * 64 + dd] = (bf16)ctx[dd];
      }
    }
    __syncthreads();

    // comb[:, :256] = bf16(xt + 0.3*(xt@ps_w + ps_b))
    gemm16<2>(af_xo, wps, bps, 256, 256, [&](int r, int c, float val) {
      float xv = xo[((size_t)r * T_ + t) * D_ + c];
      combS[r * 512 + c] = (bf16)(xv + 0.3f * val);
    });
    // comb[:, 256:] = bf16(ctxv @ ta_wo + bo), zeroed at t==0
    if (t > 0) {
      gemm16<2>([&](int r, int khalf, int kk) -> v16bf {
                  return frag_a_bf16(ctxvS + r * 256, khalf, kk);
                },
                wo, bo, 256, 256, [&](int r, int c, float val) {
                  combS[r * 512 + 256 + c] = (bf16)val;
                });
    } else {
      for (int i = tid; i < 16 * 256; i += 256) {
        int r = i >> 8, c = i & 255;
        combS[r * 512 + 256 + c] = (bf16)0.0f;
      }
    }
    __syncthreads();

    // g1 = gelu(comb @ tr_w1 + tr_b1)   (overwrites qt/ctxv region)
    gemm16<4>([&](int r, int khalf, int kk) -> v16bf {
                return frag_a_bf16(combS + r * 512, khalf, kk);
              },
              wtr1, btr1, 512, 512, [&](int r, int c, float val) {
                g1S[r * 512 + c] = (bf16)gelu_f(val);
              });
    __syncthreads();

    // delta = g1 @ tr_w2 + tr_b2   (reuses comb storage as f32)
    gemm16<2>([&](int r, int khalf, int kk) -> v16bf {
                return frag_a_bf16(g1S + r * 512, khalf, kk);
              },
              wtr2, btr2, 256, 512,
              [&](int r, int c, float val) { deltaS[r * 256 + c] = val; });
    __syncthreads();

    // h = LN3(h + 0.5*delta): 16 lanes per row, width-16 shfl reductions
    {
      const int r = tid >> 4;
      const int c0 = tid & 15;
      float vals[16];
      float sum = 0.f;
#pragma unroll
      for (int m = 0; m < 16; ++m) {
        int c = c0 + m * 16;
        float vv = hS[r * 256 + c] + 0.5f * deltaS[r * 256 + c];
        vals[m] = vv;
        sum += vv;
      }
      sum += __shfl_xor(sum, 8, 16);
      sum += __shfl_xor(sum, 4, 16);
      sum += __shfl_xor(sum, 2, 16);
      sum += __shfl_xor(sum, 1, 16);
      float mean = sum * (1.0f / 256.0f);
      float sq = 0.f;
#pragma unroll
      for (int m = 0; m < 16; ++m) {
        float d0 = vals[m] - mean;
        sq += d0 * d0;
      }
      sq += __shfl_xor(sq, 8, 16);
      sq += __shfl_xor(sq, 4, 16);
      sq += __shfl_xor(sq, 2, 16);
      sq += __shfl_xor(sq, 1, 16);
      float rstd = rsqrtf(sq * (1.0f / 256.0f) + 1e-5f);
#pragma unroll
      for (int m = 0; m < 16; ++m) {
        int c = c0 + m * 16;
        hS[r * 256 + c] = (vals[m] - mean) * rstd * ln3g[c] + ln3b[c];
      }
    }
    __syncthreads();

    // append K/V cache rows from the new h
    auto af_h = [&](int r, int khalf, int kk) -> v16bf {
      return frag_a_f32(hS + r * 256, khalf, kk);
    };
    gemm16<2>(af_h, wk, bk, 256, 256, [&](int r, int c, float val) {
      kbuf[((size_t)t * 16 + r) * 256 + c] = val;
    });
    gemm16<2>(af_h, wv, bv, 256, 256, [&](int r, int c, float val) {
      vbuf[((size_t)t * 16 + r) * 256 + c] = val;
    });
    __threadfence_block();
    __syncthreads();
  }

  // hout = LN4(hT)
  {
    const int r = tid >> 4;
    const int c0 = tid & 15;
    float vals[16];
    float sum = 0.f;
#pragma unroll
    for (int m = 0; m < 16; ++m) {
      int c = c0 + m * 16;
      vals[m] = hS[r * 256 + c];
      sum += vals[m];
    }
    sum += __shfl_xor(sum, 8, 16);
    sum += __shfl_xor(sum, 4, 16);
    sum += __shfl_xor(sum, 2, 16);
    sum += __shfl_xor(sum, 1, 16);
    float mean = sum * (1.0f / 256.0f);
    float sq = 0.f;
#pragma unroll
    for (int m = 0; m < 16; ++m) {
      float d0 = vals[m] - mean;
      sq += d0 * d0;
    }
    sq += __shfl_xor(sq, 8, 16);
    sq += __shfl_xor(sq, 4, 16);
    sq += __shfl_xor(sq, 2, 16);
    sq += __shfl_xor(sq, 1, 16);
    float rstd = rsqrtf(sq * (1.0f / 256.0f) + 1e-5f);
#pragma unroll
    for (int m = 0; m < 16; ++m) {
      int c = c0 + m * 16;
      hout[r * 256 + c] = (vals[m] - mean) * rstd * ln4g[c] + ln4b[c];
    }
  }
}

// ---------------------------------------------------------------------------
extern "C" void kernel_launch(void* const* d_in, const int* in_sizes, int n_in,
                              void* d_out, int out_size, void* d_ws,
                              size_t ws_size, hipStream_t stream) {
  (void)in_sizes; (void)n_in; (void)out_size; (void)ws_size;
  const float* x     = (const float*)d_in[0];
  const float* h0    = (const float*)d_in[1];
  const float* sa_wq = (const float*)d_in[2];
  const float* sa_wk = (const float*)d_in[3];
  const float* sa_wv = (const float*)d_in[4];
  const float* sa_wo = (const float*)d_in[5];
  const float* sa_bq = (const float*)d_in[6];
  const float* sa_bk = (const float*)d_in[7];
  const float* sa_bv = (const float*)d_in[8];
  const float* sa_bo = (const float*)d_in[9];
  const float* f_w1  = (const float*)d_in[10];
  const float* f_b1  = (const float*)d_in[11];
  const float* f_w2  = (const float*)d_in[12];
  const float* f_b2  = (const float*)d_in[13];
  const float* ta_wq = (const float*)d_in[14];
  const float* ta_wk = (const float*)d_in[15];
  const float* ta_wv = (const float*)d_in[16];
  const float* ta_wo = (const float*)d_in[17];
  const float* ta_bq = (const float*)d_in[18];
  const float* ta_bk = (const float*)d_in[19];
  const float* ta_bv = (const float*)d_in[20];
  const float* ta_bo = (const float*)d_in[21];
  const float* tr_w1 = (const float*)d_in[22];
  const float* tr_b1 = (const float*)d_in[23];
  const float* tr_w2 = (const float*)d_in[24];
  const float* tr_b2 = (const float*)d_in[25];
  const float* ps_w  = (const float*)d_in[26];
  const float* ps_b  = (const float*)d_in[27];
  const float* sp_w  = (const float*)d_in[28];
  const float* sp_b  = (const float*)d_in[29];
  const float* ln1_g = (const float*)d_in[30];
  const float* ln1_b = (const float*)d_in[31];
  const float* ln2_g = (const float*)d_in[32];
  const float* ln2_b = (const float*)d_in[33];
  const float* ln3_g = (const float*)d_in[34];
  const float* ln3_b = (const float*)d_in[35];
  const float* ln4_g = (const float*)d_in[36];
  const float* ln4_b = (const float*)d_in[37];

  float* xout = (float*)d_out;                         // [B,T,D]
  float* hout = (float*)d_out + (size_t)B_ * T_ * D_;  // [B,D]

  char* base = (char*)d_ws;
  size_t off = 0;
  auto falloc = [&](size_t n) -> float* {
    float* p = (float*)(base + off);
    off += n * 4;
    return p;
  };
  auto balloc = [&](size_t n) -> bf16* {
    bf16* p = (bf16*)(base + off);
    off += ((n * 2 + 15) & ~(size_t)15);
    return p;
  };
  const size_t MT = (size_t)B_ * T_;  // 8192
  float* attn = falloc(MT * D_);
  float* t0   = falloc(MT * D_);
  float* x1   = falloc(MT * D_);
  float* ffin = falloc(MT * D_);
  float* h1m  = falloc(MT * 4 * D_);
  float* t1   = falloc(MT * D_);
  float* guid = falloc((size_t)B_ * D_);
  float* kbuf = falloc((size_t)T_ * B_ * D_);
  float* vbuf = falloc((size_t)T_ * B_ * D_);
  bf16* qbb   = balloc(MT * D_);
  bf16* kbb   = balloc(MT * D_);
  bf16* vtb   = balloc(MT * D_);  // V^T [b][h][d][t]
  bf16* bwq   = balloc(D_ * D_);
  bf16* bwk   = balloc(D_ * D_);
  bf16* bwv   = balloc(D_ * D_);
  bf16* bwo   = balloc(D_ * D_);
  bf16* bwps  = balloc(D_ * D_);
  bf16* bwtr1 = balloc(2 * D_ * 2 * D_);
  bf16* bwtr2 = balloc(2 * D_ * D_);

  // one-time transposed bf16 conversion of scan weights (L2-resident after)
  auto cvt_t = [&](const float* s, bf16* d, int K, int N) {
    cvt_t_kernel<<<(K * N + 255) / 256, 256, 0, stream>>>(s, d, K, N);
  };
  cvt_t(ta_wq, bwq, 256, 256);
  cvt_t(ta_wk, bwk, 256, 256);
  cvt_t(ta_wv, bwv, 256, 256);
  cvt_t(ta_wo, bwo, 256, 256);
  cvt_t(ps_w, bwps, 256, 256);
  cvt_t(tr_w1, bwtr1, 512, 512);
  cvt_t(tr_w2, bwtr2, 512, 256);

  const dim3 blk(256);
  // ---- parallel branch ----
  gemm_kernel<0, false, 1><<<dim3(4, 64), blk, 0, stream>>>(
      x, sa_wq, sa_bq, nullptr, nullptr, qbb, nullptr, 8192, 256, 256);
  gemm_kernel<0, false, 1><<<dim3(4, 64), blk, 0, stream>>>(
      x, sa_wk, sa_bk, nullptr, nullptr, kbb, nullptr, 8192, 256, 256);
  gemm_kernel<0, false, 2><<<dim3(4, 64), blk, 0, stream>>>(
      x, sa_wv, sa_bv, nullptr, nullptr, nullptr, vtb, 8192, 256, 256);
  attn_kernel<<<dim3(T_ / 16, H_, B_), dim3(32), 0, stream>>>(qbb, kbb, vtb, attn);
  gemm_kernel<0, true, 0><<<dim3(4, 64), blk, 0, stream>>>(
      attn, sa_wo, sa_bo, x, t0, nullptr, nullptr, 8192, 256, 256);
  ln_kernel<<<8192, 256, 0, stream>>>(t0, ln1_g, ln1_b, x1);
  gemm_kernel<0, false, 0><<<dim3(4, 1), blk, 0, stream>>>(
      h0, sp_w, sp_b, nullptr, guid, nullptr, nullptr, 16, 256, 256);
  add_guid_kernel<<<8192, 256, 0, stream>>>(x1, guid, ffin);
  gemm_kernel<1, false, 0><<<dim3(16, 64), blk, 0, stream>>>(
      ffin, f_w1, f_b1, nullptr, h1m, nullptr, nullptr, 8192, 1024, 256);
  gemm_kernel<0, true, 0><<<dim3(4, 64), blk, 0, stream>>>(
      h1m, f_w2, f_b2, x1, t1, nullptr, nullptr, 8192, 256, 1024);
  ln_kernel<<<8192, 256, 0, stream>>>(t1, ln2_g, ln2_b, xout);

  // ---- sequential branch (persistent single workgroup) ----
  scan_kernel<<<1, 256, 0, stream>>>(xout, h0, bwq, bwk, bwv, bwo, bwps, bwtr1,
                                     bwtr2, ta_bq, ta_bk, ta_bv, ta_bo, ps_b,
                                     tr_b1, tr_b2, ln3_g, ln3_b, ln4_g, ln4_b,
                                     kbuf, vbuf, hout);
}
